// TransformerBlock_39642548142316
// MI455X (gfx1250) — matrix operand order repaired and checked by reading, not compiled, not run on hardware
//
#include <hip/hip_runtime.h>

// ---------------------------------------------------------------------------
// Transformer block for MI455X (gfx1250, wave32, WMMA).
// - All matmuls via v_wmma_f32_16x16x32_bf16 (fp32 accumulation).
// - GEMM: 256x128 block tile, 64x64 per wave (1 ds_load_b128 per WMMA),
//   double-buffered LDS fed by global_load_async_to_lds_b128 (ASYNCcnt).
// - Attention: causal flash-style (online softmax), async K/Q tile staging.
// Workspace use: ~120 MB.
// ---------------------------------------------------------------------------

typedef __attribute__((ext_vector_type(16))) __bf16        v16bf;
typedef __attribute__((ext_vector_type(8)))  float         v8f;
typedef __attribute__((ext_vector_type(8)))  unsigned short us8;

union Frag {
  v16bf bf;
  us8   u[2];
};

__device__ __forceinline__ unsigned short f2bf(float x) {
  unsigned u = __float_as_uint(x);
  u += 0x7FFFu + ((u >> 16) & 1u);   // round-to-nearest-even
  return (unsigned short)(u >> 16);
}

__device__ __forceinline__ v8f wmma_bf16(const Frag& a, const Frag& b, v8f c) {
  return __builtin_amdgcn_wmma_f32_16x16x32_bf16(false, a.bf, false, b.bf,
                                                 (short)0, c, false, false);
}

// Async global -> LDS copy of 16 bytes per lane (tracked by ASYNCcnt).
// VDST = per-lane LDS byte offset (low 32 bits of the flat shared address),
// VADDR = per-lane 64-bit global address, GV mode (saddr = off).
__device__ __forceinline__ void async_b128(void* lds_ptr, const void* gptr) {
  unsigned lds_off = (unsigned)(size_t)lds_ptr;
  unsigned long long ga = (unsigned long long)(size_t)gptr;
  asm volatile("global_load_async_to_lds_b128 %0, %1, off"
               :: "v"(lds_off), "v"(ga) : "memory");
}
__device__ __forceinline__ void wait_async0() {
  asm volatile("s_wait_asynccnt 0x0" ::: "memory");
}

// -------------------------------- fp32 -> bf16 ------------------------------
__global__ void cvt_bf16_kernel(const float* __restrict__ in,
                                unsigned short* __restrict__ out, int n) {
  int i = blockIdx.x * 256 + threadIdx.x;
  if (i < n) out[i] = f2bf(in[i]);
}

// -------------------------------- GEMM --------------------------------------
// C[m,n] = sum_k A[m,k] * B[n,k]   (A: MxK bf16, B: NxK bf16, both row-major)
// Optional: + bias[n], + resid[m,n] (f32), ReLU; writes f32 and/or bf16.
// Block: 256 threads (8 waves). Tile 256x128, k-step 32. Wave -> 64x64.
__global__ __launch_bounds__(256)
void gemm_bf16_kernel(const unsigned short* __restrict__ A,
                      const unsigned short* __restrict__ Bw,
                      int M, int N, int K,
                      const float* __restrict__ bias,
                      const float* __restrict__ resid,
                      float* __restrict__ outF,
                      unsigned short* __restrict__ outB,
                      int relu) {
  __shared__ __align__(16) unsigned short As[2][256][40];  // 32 + 8 pad
  __shared__ __align__(16) unsigned short Bs[2][128][40];

  const int tid  = threadIdx.x;
  const int wid  = tid >> 5;
  const int lane = tid & 31;
  const int half = lane >> 4;
  const int ln   = lane & 15;
  const int wm   = wid >> 1;            // 0..3 -> 64-row strip
  const int wn   = wid & 1;             // 0..1 -> 64-col strip
  const int tm0  = blockIdx.y * 256;
  const int tn0  = blockIdx.x * 128;

  v8f acc[4][4];
  v8f zero = {0.f, 0.f, 0.f, 0.f, 0.f, 0.f, 0.f, 0.f};
#pragma unroll
  for (int i = 0; i < 4; i++)
#pragma unroll
    for (int j = 0; j < 4; j++) acc[i][j] = zero;

  // issue one 256x32 A tile + 128x32 B tile as async b128 copies
  auto issue_tile = [&](int buf, int kt) {
#pragma unroll
    for (int j = 0; j < 4; j++) {                 // A: 1024 chunks / 256 thr
      int c = j * 256 + tid;
      int row = c >> 2, seg = (c & 3) * 8;
      async_b128(&As[buf][row][seg],
                 A + (size_t)(tm0 + row) * K + kt + seg);
    }
#pragma unroll
    for (int j = 0; j < 2; j++) {                 // B: 512 chunks / 256 thr
      int c = j * 256 + tid;
      int row = c >> 2, seg = (c & 3) * 8;
      async_b128(&Bs[buf][row][seg],
                 Bw + (size_t)(tn0 + row) * K + kt + seg);
    }
  };

  const int nk = K >> 5;
  issue_tile(0, 0);

  for (int ki = 0; ki < nk; ki++) {
    const int cur = ki & 1;
    wait_async0();
    __syncthreads();                    // buf[cur] ready; buf[cur^1] free
    if (ki + 1 < nk) issue_tile(cur ^ 1, (ki + 1) << 5);

    Frag a[4];
#pragma unroll
    for (int mt = 0; mt < 4; mt++) {
      int m = wm * 64 + mt * 16 + ln;
      a[mt].u[0] = *(const us8*)(&As[cur][m][8 * half]);
      a[mt].u[1] = *(const us8*)(&As[cur][m][16 + 8 * half]);
    }
#pragma unroll
    for (int nt = 0; nt < 4; nt++) {
      Frag b;
      int nr = wn * 64 + nt * 16 + ln;
      b.u[0] = *(const us8*)(&Bs[cur][nr][8 * half]);
      b.u[1] = *(const us8*)(&Bs[cur][nr][16 + 8 * half]);
#pragma unroll
      for (int mt = 0; mt < 4; mt++)
        acc[mt][nt] = wmma_bf16(a[mt], b, acc[mt][nt]);
    }
  }

#pragma unroll
  for (int mt = 0; mt < 4; mt++) {
#pragma unroll
    for (int nt = 0; nt < 4; nt++) {
#pragma unroll
      for (int r = 0; r < 8; r++) {
        int m = tm0 + wm * 64 + mt * 16 + r + 8 * half;
        int n = tn0 + wn * 64 + nt * 16 + ln;
        float v = acc[mt][nt][r];
        if (bias)  v += bias[n];
        if (resid) v += resid[(size_t)m * N + n];
        if (relu)  v = v > 0.f ? v : 0.f;
        if (outF)  outF[(size_t)m * N + n] = v;
        if (outB)  outB[(size_t)m * N + n] = f2bf(v);
      }
    }
  }
}

// ----------------------------- Flash attention ------------------------------
// Q/K/V: bf16 [B*S, H*64] (feature col = head*64 + h). Z: same layout, bf16.
// Grid: (S/128, H, B). Block 256 = 8 waves; each wave owns 16 query rows.
// Causal; streams 128-key blocks with online softmax; scale 1/sqrt(64)=0.125.
#define SEQ_LEN 2048

__global__ __launch_bounds__(256)
void flash_attn_kernel(const unsigned short* __restrict__ Q,
                       const unsigned short* __restrict__ Kk,
                       const unsigned short* __restrict__ V,
                       unsigned short* __restrict__ Z) {
  __shared__ __align__(16) unsigned short Qs[128][72];       // [query][dim]
  __shared__ __align__(16) unsigned short Ks[128][72];       // [key][dim]
  __shared__ __align__(16) unsigned short Vt[64][136];       // [dim][key]
  __shared__ __align__(16) unsigned short Ps[8][16][136];    // per-wave P

  const int tid  = threadIdx.x;
  const int wid  = tid >> 5;
  const int lane = tid & 31;
  const int half = lane >> 4;
  const int ln   = lane & 15;

  const int qb    = blockIdx.x;          // query block 0..15
  const int head  = blockIdx.y;
  const int batch = blockIdx.z;
  const size_t rowbase = (size_t)batch * SEQ_LEN;
  const int fcol = head * 64;

  // async-stage the Q tile (128 x 64 = 1024 b128 chunks)
#pragma unroll
  for (int j = 0; j < 4; j++) {
    int c = j * 256 + tid;
    int row = c >> 3, seg = (c & 7) * 8;
    async_b128(&Qs[row][seg],
               Q + (rowbase + qb * 128 + row) * 1024 + fcol + seg);
  }

  const int lr   = tid >> 1;             // 0..127
  const int lseg = (tid & 1) * 32;       // dim half 0 / 32

  float mrow[8], lrow[8];
#pragma unroll
  for (int r = 0; r < 8; r++) { mrow[r] = -3.0e38f; lrow[r] = 0.f; }
  v8f zero = {0.f, 0.f, 0.f, 0.f, 0.f, 0.f, 0.f, 0.f};
  v8f Oacc[4];
#pragma unroll
  for (int d = 0; d < 4; d++) Oacc[d] = zero;

  for (int kb = 0; kb <= qb; kb++) {
    __syncthreads();                 // all waves done reading Ks/Vt
    // K tile: async row-major copy
#pragma unroll
    for (int j = 0; j < 4; j++) {
      int c = j * 256 + tid;
      int row = c >> 3, seg = (c & 7) * 8;
      async_b128(&Ks[row][seg],
                 Kk + (rowbase + kb * 128 + row) * 1024 + fcol + seg);
    }
    // V tile: register round-trip with transpose into Vt[dim][key]
    {
      const unsigned short* sv = V + (rowbase + kb * 128 + lr) * 1024 + fcol + lseg;
      unsigned short tmp[32];
#pragma unroll
      for (int j = 0; j < 4; j++)
        *(us8*)(&tmp[j * 8]) = *(const us8*)(sv + j * 8);
#pragma unroll
      for (int j = 0; j < 32; j++)
        Vt[lseg + j][lr] = tmp[j];
    }
    wait_async0();
    __syncthreads();

    // S = Q * K^T : wave's 16 rows x 128 keys
    v8f Sacc[8];
#pragma unroll
    for (int nt = 0; nt < 8; nt++) Sacc[nt] = zero;
#pragma unroll
    for (int ktp = 0; ktp < 2; ktp++) {
      Frag a;
      int m = wid * 16 + ln;
      a.u[0] = *(const us8*)(&Qs[m][ktp * 32 + 8 * half]);
      a.u[1] = *(const us8*)(&Qs[m][ktp * 32 + 16 + 8 * half]);
#pragma unroll
      for (int nt = 0; nt < 8; nt++) {
        Frag b;
        int kr = nt * 16 + ln;
        b.u[0] = *(const us8*)(&Ks[kr][ktp * 32 + 8 * half]);
        b.u[1] = *(const us8*)(&Ks[kr][ktp * 32 + 16 + 8 * half]);
        Sacc[nt] = wmma_bf16(a, b, Sacc[nt]);
      }
    }

    // scale + causal mask
    const int q0 = qb * 128 + wid * 16;
#pragma unroll
    for (int nt = 0; nt < 8; nt++) {
      int p = kb * 128 + nt * 16 + ln;
#pragma unroll
      for (int r = 0; r < 8; r++) {
        int qrow = q0 + r + 8 * half;
        float s = Sacc[nt][r] * 0.125f;
        if (p > qrow) s = -3.0e38f;
        Sacc[nt][r] = s;
      }
    }

    // new row max (xor 1,2,4,8 stays inside each 16-lane half)
    float mnew[8];
#pragma unroll
    for (int r = 0; r < 8; r++) {
      float mx = mrow[r];
#pragma unroll
      for (int nt = 0; nt < 8; nt++) mx = fmaxf(mx, Sacc[nt][r]);
      mx = fmaxf(mx, __shfl_xor(mx, 1, 32));
      mx = fmaxf(mx, __shfl_xor(mx, 2, 32));
      mx = fmaxf(mx, __shfl_xor(mx, 4, 32));
      mx = fmaxf(mx, __shfl_xor(mx, 8, 32));
      mnew[r] = mx;
    }

    // exponentiate, stash P (bf16) in LDS, row-sum
    float rsum[8];
#pragma unroll
    for (int r = 0; r < 8; r++) rsum[r] = 0.f;
#pragma unroll
    for (int nt = 0; nt < 8; nt++) {
#pragma unroll
      for (int r = 0; r < 8; r++) {
        float pv = __expf(Sacc[nt][r] - mnew[r]);
        rsum[r] += pv;
        Ps[wid][r + 8 * half][nt * 16 + ln] = f2bf(pv);
      }
    }
#pragma unroll
    for (int r = 0; r < 8; r++) {
      float s = rsum[r];
      s += __shfl_xor(s, 1, 32);
      s += __shfl_xor(s, 2, 32);
      s += __shfl_xor(s, 4, 32);
      s += __shfl_xor(s, 8, 32);
      float alpha = __expf(mrow[r] - mnew[r]);
      lrow[r] = lrow[r] * alpha + s;
      mrow[r] = mnew[r];
#pragma unroll
      for (int d = 0; d < 4; d++) Oacc[d][r] *= alpha;
    }
    __syncthreads();   // uniform per-block (kb bound depends only on blockIdx)

    // O += P * V  (A = P 16x128, B = V 128x64 via transposed Vt)
#pragma unroll
    for (int ktp = 0; ktp < 4; ktp++) {
      Frag a;
      a.u[0] = *(const us8*)(&Ps[wid][ln][ktp * 32 + 8 * half]);
      a.u[1] = *(const us8*)(&Ps[wid][ln][ktp * 32 + 16 + 8 * half]);
#pragma unroll
      for (int d = 0; d < 4; d++) {
        Frag b;
        int dn = d * 16 + ln;
        b.u[0] = *(const us8*)(&Vt[dn][ktp * 32 + 8 * half]);
        b.u[1] = *(const us8*)(&Vt[dn][ktp * 32 + 16 + 8 * half]);
        Oacc[d] = wmma_bf16(a, b, Oacc[d]);
      }
    }
  }

  // normalize and write Z (bf16), layout [b*S + q][head*64 + h]
#pragma unroll
  for (int d = 0; d < 4; d++) {
#pragma unroll
    for (int r = 0; r < 8; r++) {
      size_t row = rowbase + qb * 128 + wid * 16 + r + 8 * half;
      Z[row * 1024 + fcol + d * 16 + ln] = f2bf(Oacc[d][r] / lrow[r]);
    }
  }
}

// ------------------------------- launcher -----------------------------------
extern "C" void kernel_launch(void* const* d_in, const int* in_sizes, int n_in,
                              void* d_out, int out_size, void* d_ws, size_t ws_size,
                              hipStream_t stream) {
  (void)in_sizes; (void)n_in; (void)out_size; (void)ws_size;

  const float* x    = (const float*)d_in[0];
  const float* W_K  = (const float*)d_in[1];
  const float* W_Q  = (const float*)d_in[2];
  const float* W_V  = (const float*)d_in[3];
  const float* W_O  = (const float*)d_in[4];
  const float* W_in = (const float*)d_in[5];
  const float* b_in = (const float*)d_in[6];
  const float* W_out= (const float*)d_in[7];
  const float* b_out= (const float*)d_in[8];
  float* out = (float*)d_out;

  const int Mr = 4096;   // B*S
  const int D  = 1024;
  const int DM = 4096;

  char* ws = (char*)d_ws;
  size_t off = 0;
  auto alloc = [&](size_t bytes) -> char* {
    char* p = ws + off;
    off += (bytes + 255) & ~(size_t)255;
    return p;
  };
  unsigned short* xb    = (unsigned short*)alloc((size_t)Mr * D * 2);   //  8 MB
  unsigned short* wqb   = (unsigned short*)alloc((size_t)D * D * 2);    //  2 MB
  unsigned short* wkb   = (unsigned short*)alloc((size_t)D * D * 2);    //  2 MB
  unsigned short* wvb   = (unsigned short*)alloc((size_t)D * D * 2);    //  2 MB
  unsigned short* wob   = (unsigned short*)alloc((size_t)D * D * 2);    //  2 MB
  unsigned short* winb  = (unsigned short*)alloc((size_t)DM * D * 2);   //  8 MB
  unsigned short* woutb = (unsigned short*)alloc((size_t)D * DM * 2);   //  8 MB
  unsigned short* qbuf  = (unsigned short*)alloc((size_t)Mr * D * 2);   //  8 MB
  unsigned short* kbuf  = (unsigned short*)alloc((size_t)Mr * D * 2);   //  8 MB
  unsigned short* vbuf  = (unsigned short*)alloc((size_t)Mr * D * 2);   //  8 MB
  unsigned short* zb    = (unsigned short*)alloc((size_t)Mr * D * 2);   //  8 MB
  float*          x1    = (float*)         alloc((size_t)Mr * D * 4);   // 16 MB
  unsigned short* x1b   = (unsigned short*)alloc((size_t)Mr * D * 2);   //  8 MB
  unsigned short* hb    = (unsigned short*)alloc((size_t)Mr * DM * 2);  // 32 MB
  // total ~120 MB of workspace

  auto cvt = [&](const float* src, unsigned short* dst, int n) {
    cvt_bf16_kernel<<<(n + 255) / 256, 256, 0, stream>>>(src, dst, n);
  };
  cvt(x,     xb,    Mr * D);
  cvt(W_Q,   wqb,   D * D);
  cvt(W_K,   wkb,   D * D);
  cvt(W_V,   wvb,   D * D);
  cvt(W_O,   wob,   D * D);
  cvt(W_in,  winb,  DM * D);
  cvt(W_out, woutb, D * DM);

  // QKV projections: [4096,1024] x [1024,1024]^T -> bf16
  gemm_bf16_kernel<<<dim3(D / 128, Mr / 256), 256, 0, stream>>>(
      xb, wqb, Mr, D, D, nullptr, nullptr, nullptr, qbuf, 0);
  gemm_bf16_kernel<<<dim3(D / 128, Mr / 256), 256, 0, stream>>>(
      xb, wkb, Mr, D, D, nullptr, nullptr, nullptr, kbuf, 0);
  gemm_bf16_kernel<<<dim3(D / 128, Mr / 256), 256, 0, stream>>>(
      xb, wvb, Mr, D, D, nullptr, nullptr, nullptr, vbuf, 0);

  // causal flash attention -> zb
  flash_attn_kernel<<<dim3(SEQ_LEN / 128, 16, 2), 256, 0, stream>>>(
      qbuf, kbuf, vbuf, zb);

  // attn_out = Z * W_O^T ; x1 = x + attn_out (f32 + bf16 copies)
  gemm_bf16_kernel<<<dim3(D / 128, Mr / 256), 256, 0, stream>>>(
      zb, wob, Mr, D, D, nullptr, x, x1, x1b, 0);

  // h = relu(x1 * W_in^T + b_in) -> bf16
  gemm_bf16_kernel<<<dim3(DM / 128, Mr / 256), 256, 0, stream>>>(
      x1b, winb, Mr, DM, D, b_in, nullptr, nullptr, hb, 1);

  // out = x1 + h * W_out^T + b_out -> f32
  gemm_bf16_kernel<<<dim3(D / 128, Mr / 256), 256, 0, stream>>>(
      hb, woutb, Mr, D, DM, b_out, x1, out, nullptr, 0);
}
